// SequenceLayer_8177617731970
// MI455X (gfx1250) — matrix operand order repaired and checked by reading, not compiled, not run on hardware
//
#include <hip/hip_runtime.h>

// ---------------------------------------------------------------------------
// Types for CDNA5 WMMA / TDM
// ---------------------------------------------------------------------------
typedef __attribute__((ext_vector_type(16))) __bf16       v16bf;
typedef __attribute__((ext_vector_type(8)))  float        v8f;
typedef __attribute__((ext_vector_type(4)))  unsigned int u32x4;
typedef __attribute__((ext_vector_type(4)))  int          i32x4;
typedef __attribute__((ext_vector_type(8)))  int          i32x8;

#define S_DIM   2048
#define B_DIM   8
#define M_TOTAL (B_DIM * S_DIM)        /* 16384 rows of x                     */
#define MT_BLK  64                     /* M rows per block                    */
#define NT_BLK  64                     /* N cols (per gate) per block         */
#define KC      64                     /* K chunk per TDM stage               */
#define STAGES  (S_DIM / KC)           /* 32                                  */
#define A_HALVES (MT_BLK * KC)         /* 4096 bf16 = 8 KB                    */
#define B_HALVES (3 * NT_BLK * KC)     /* 12288 bf16 = 24 KB                  */
#define STAGE_HALVES (A_HALVES + B_HALVES)   /* 16384 = 32 KB; x2 = 64 KB LDS */

// Memory-clobber fence: with the LDS base escaped into an asm (see below),
// this is a potential writer of the LDS tile, so staged loads cannot be
// folded away even though only the TDM (invisible to the compiler) writes it.
#define LDS_FENCE() asm volatile("" ::: "memory")

union FragBF { u32x4 u[2]; v16bf v; };

__device__ __forceinline__ unsigned short f2bf(float f) {
  unsigned u = __float_as_uint(f);
  u += 0x7FFFu + ((u >> 16) & 1u);   // round-to-nearest-even
  return (unsigned short)(u >> 16);
}

__device__ __forceinline__ float fast_sigmoid(float x) {
  return 1.0f / (1.0f + __expf(-x));
}
__device__ __forceinline__ float fast_tanh(float x) {
  return 1.0f - 2.0f / (__expf(2.0f * x) + 1.0f);
}

// ---------------------------------------------------------------------------
// TDM issue: up-to-3D tile load, descriptor packed per CDNA5 ISA ch.8.
// dim0 = contiguous axis, dim1 strided by d0_stride, dim2 strided by d1_stride.
// ---------------------------------------------------------------------------
__device__ __forceinline__ void tdm_issue(const void* gptr, unsigned lds_byte,
                                          unsigned tensor_d0, unsigned tensor_d1,
                                          unsigned tile_d0, unsigned tile_d1,
                                          unsigned tile_d2, unsigned tensor_d2,
                                          unsigned long long d0_stride,
                                          unsigned long long d1_stride) {
  unsigned long long ga = (unsigned long long)gptr;
  // group0: count=1 | lds_addr | global_addr(57b) | type=2
  u32x4 g0 = { 1u, lds_byte, (unsigned)ga,
               (unsigned)((ga >> 32) & 0x01FFFFFFu) | 0x80000000u };
  // group1: data_size=1 (2B elements); dims/tiles/strides per bit layout
  i32x8 g1 = { (int)(1u << 16),
               (int)((tensor_d0 & 0xFFFFu) << 16),
               (int)((tensor_d0 >> 16) | ((tensor_d1 & 0xFFFFu) << 16)),
               (int)((tensor_d1 >> 16) | (tile_d0 << 16)),
               (int)(tile_d1 | (tile_d2 << 16)),
               (int)(unsigned)(d0_stride & 0xFFFFFFFFu),
               (int)((unsigned)((d0_stride >> 32) & 0xFFFFu) |
                     (unsigned)((d1_stride & 0xFFFFu) << 16)),
               (int)(unsigned)((d1_stride >> 16) & 0xFFFFFFFFu) };
  // group2: tensor_dim2 | tensor_dim3=0 | dim2_stride=0 | tile_dim3=0
  i32x4 g2 = { (int)tensor_d2, 0, 0, 0 };
  i32x4 g3 = { 0, 0, 0, 0 };
#if __has_include(<hip/amd_detail/amd_gfx1250_TDM.h>)
  i32x8 z8 = { 0, 0, 0, 0, 0, 0, 0, 0 };
  __builtin_amdgcn_tensor_load_to_lds(g0, g1, g2, g3, z8, 0);
#else
  __builtin_amdgcn_tensor_load_to_lds(g0, g1, g2, g3, 0);
#endif
}

// ---------------------------------------------------------------------------
// Kernel 1: f32 -> bf16 conversion of x (33.5M elements), 8 per thread
// ---------------------------------------------------------------------------
__global__ __launch_bounds__(256) void cvt_x_kernel(const float* __restrict__ in,
                                                    unsigned short* __restrict__ out) {
  size_t base = ((size_t)blockIdx.x * 256 + threadIdx.x) * 8;
  float4 a = *(const float4*)(in + base);
  float4 b = *(const float4*)(in + base + 4);
  u32x4 p = { (unsigned)f2bf(a.x) | ((unsigned)f2bf(a.y) << 16),
              (unsigned)f2bf(a.z) | ((unsigned)f2bf(a.w) << 16),
              (unsigned)f2bf(b.x) | ((unsigned)f2bf(b.y) << 16),
              (unsigned)f2bf(b.z) | ((unsigned)f2bf(b.w) << 16) };
  *(u32x4*)(out + base) = p;
}

// ---------------------------------------------------------------------------
// Kernel 2: f32 -> bf16 for the i, g, o blocks of w_ih (f block unused: c0=0)
// ---------------------------------------------------------------------------
__global__ __launch_bounds__(256) void cvt_w_kernel(const float* __restrict__ w,
                                                    unsigned short* __restrict__ out) {
  size_t j     = ((size_t)blockIdx.x * 256 + threadIdx.x) * 8;
  size_t which = j / ((size_t)S_DIM * S_DIM);          // 0,1,2 -> gate blocks i,g,o
  size_t off   = j % ((size_t)S_DIM * S_DIM);
  size_t sel   = (which == 0) ? 0 : (which + 1);       // 0 -> 0, 1 -> 2, 2 -> 3
  const float* src = w + sel * (size_t)S_DIM * S_DIM + off;
  float4 a = *(const float4*)(src);
  float4 b = *(const float4*)(src + 4);
  u32x4 p = { (unsigned)f2bf(a.x) | ((unsigned)f2bf(a.y) << 16),
              (unsigned)f2bf(a.z) | ((unsigned)f2bf(a.w) << 16),
              (unsigned)f2bf(b.x) | ((unsigned)f2bf(b.y) << 16),
              (unsigned)f2bf(b.z) | ((unsigned)f2bf(b.w) << 16) };
  *(u32x4*)(out + j) = p;
}

// ---------------------------------------------------------------------------
// Kernel 3: tiled GEMM (3 gate matrices) + LSTM-cell nonlinearity.
// Grid: (M_TOTAL/64, S/64). Block = 256 threads = 8 waves (2 x 4).
// Double-buffered LDS stages filled by the TDM:
//   A chunk : 64 rows x 64 K  (2D descriptor)
//   B chunk : 3 gates x 64 cols x 64 K in ONE 3D descriptor
//             (tile_dim2 = 3, tensor_dim1_stride = S*S)
// Each wave: two 16x16 M-subtiles x {i,g,o} accumulators, 6 WMMAs per k-step.
// ---------------------------------------------------------------------------
__global__ __launch_bounds__(256) void gemm_gates_kernel(const unsigned short* __restrict__ A,
                                                         const unsigned short* __restrict__ W,
                                                         float* __restrict__ out) {
  __shared__ unsigned short smem[2 * STAGE_HALVES];   // 64 KB, LDS offset 0

  // Escape the LDS base into a memory-clobbering asm: the compiler must now
  // assume asm/fences can write smem, so it cannot fold the staged ds_loads
  // to undef (only the TDM hardware actually writes this memory).
  asm volatile("" :: "v"((const void*)smem) : "memory");

  const int tid    = threadIdx.x;
  const int wave   = tid >> 5;
  const int lane   = tid & 31;
  const int laneN  = lane & 15;
  const int laneHi = lane >> 4;
  const int wm     = wave >> 2;          // 0..1  -> M 32-row half
  const int wn     = wave & 3;           // 0..3  -> N 16-col slice
  const size_t row0 = (size_t)blockIdx.x * MT_BLK;
  const size_t n0   = (size_t)blockIdx.y * NT_BLK;

  // Fragment lane layout (ISA 7.12.2, 16-bit):
  //   A: lane<16 holds K {0..7,16..23}; lane>=16 holds K {8..15,24..31}
  //   B: lane<16 holds K {0..7,16..23}; lane>=16 holds K {8..15,24..31}
  const int ka = laneHi * 8;
  const int kb = laneHi * 8;

  // ---- prologue: stage 0
  if (wave == 0) {
    tdm_issue(A + row0 * S_DIM, 0u,
              S_DIM, M_TOTAL, KC, MT_BLK, 0, 0, S_DIM, 0);
    tdm_issue(W + n0 * S_DIM, (unsigned)(A_HALVES * 2),
              S_DIM, S_DIM, KC, NT_BLK, 3, 3,
              S_DIM, (unsigned long long)S_DIM * S_DIM);
    __builtin_amdgcn_s_wait_tensorcnt(0);
  }
  __syncthreads();
  LDS_FENCE();

  v8f acc_i0 = {}, acc_g0 = {}, acc_o0 = {};
  v8f acc_i1 = {}, acc_g1 = {}, acc_o1 = {};

  for (int s = 0; s < STAGES; ++s) {
    // ---- issue next stage's DMA into the other buffer
    if (wave == 0 && s + 1 < STAGES) {
      const int k1 = (s + 1) * KC;
      const unsigned base = (unsigned)(((s + 1) & 1) * STAGE_HALVES * 2);
      tdm_issue(A + row0 * S_DIM + k1, base,
                S_DIM, M_TOTAL, KC, MT_BLK, 0, 0, S_DIM, 0);
      tdm_issue(W + n0 * S_DIM + k1, base + (unsigned)(A_HALVES * 2),
                S_DIM, S_DIM, KC, NT_BLK, 3, 3,
                S_DIM, (unsigned long long)S_DIM * S_DIM);
    }

    // ---- compute on current buffer
    const unsigned short* sa = smem + (size_t)(s & 1) * STAGE_HALVES;
    const unsigned short* sb = sa + A_HALVES;
    const unsigned short* arow0 = sa + (size_t)(wm * 32 + laneN) * KC;      // subtile 0
    const unsigned short* arow1 = arow0 + 16 * KC;                          // subtile 1
    const unsigned short* brow  = sb + (size_t)(wn * 16 + laneN) * KC;      // gate i

#pragma unroll
    for (int ks = 0; ks < KC; ks += 32) {
      FragBF a0, a1, bi, bg, bo;
      a0.u[0] = *(const u32x4*)(arow0 + ks + ka);          // ds_load_b128
      a0.u[1] = *(const u32x4*)(arow0 + ks + ka + 16);
      a1.u[0] = *(const u32x4*)(arow1 + ks + ka);
      a1.u[1] = *(const u32x4*)(arow1 + ks + ka + 16);
      bi.u[0] = *(const u32x4*)(brow + ks + kb);
      bi.u[1] = *(const u32x4*)(brow + ks + kb + 16);
      bg.u[0] = *(const u32x4*)(brow + NT_BLK * KC + ks + kb);
      bg.u[1] = *(const u32x4*)(brow + NT_BLK * KC + ks + kb + 16);
      bo.u[0] = *(const u32x4*)(brow + 2 * NT_BLK * KC + ks + kb);
      bo.u[1] = *(const u32x4*)(brow + 2 * NT_BLK * KC + ks + kb + 16);
      acc_i0 = __builtin_amdgcn_wmma_f32_16x16x32_bf16(false, a0.v, false, bi.v,
                                                       (short)0, acc_i0, false, false);
      acc_i1 = __builtin_amdgcn_wmma_f32_16x16x32_bf16(false, a1.v, false, bi.v,
                                                       (short)0, acc_i1, false, false);
      acc_g0 = __builtin_amdgcn_wmma_f32_16x16x32_bf16(false, a0.v, false, bg.v,
                                                       (short)0, acc_g0, false, false);
      acc_g1 = __builtin_amdgcn_wmma_f32_16x16x32_bf16(false, a1.v, false, bg.v,
                                                       (short)0, acc_g1, false, false);
      acc_o0 = __builtin_amdgcn_wmma_f32_16x16x32_bf16(false, a0.v, false, bo.v,
                                                       (short)0, acc_o0, false, false);
      acc_o1 = __builtin_amdgcn_wmma_f32_16x16x32_bf16(false, a1.v, false, bo.v,
                                                       (short)0, acc_o1, false, false);
    }

    // ---- stage boundary: DMA of next stage done + all waves done reading
    if (wave == 0) __builtin_amdgcn_s_wait_tensorcnt(0);
    __syncthreads();
    LDS_FENCE();
  }

  // ---- epilogue: LSTM gates + ALPHA scale; softmax runs as a second kernel.
  // C/D layout: lane L, VGPR r -> row = r + (L>=16 ? 8 : 0), col = L%16
  const size_t col = n0 + (size_t)wn * 16 + laneN;
#pragma unroll
  for (int r = 0; r < 8; ++r) {
    size_t rbase = row0 + (size_t)wm * 32 + (size_t)r + (size_t)laneHi * 8;
    {
      float c = fast_sigmoid(acc_i0[r]) * fast_tanh(acc_g0[r]);
      float h = fast_sigmoid(acc_o0[r]) * fast_tanh(c);
      out[rbase * S_DIM + col] = 20.0f * h;
    }
    {
      float c = fast_sigmoid(acc_i1[r]) * fast_tanh(acc_g1[r]);
      float h = fast_sigmoid(acc_o1[r]) * fast_tanh(c);
      out[(rbase + 16) * S_DIM + col] = 20.0f * h;
    }
  }
}

// ---------------------------------------------------------------------------
// Kernel 4: in-place row softmax over 2048 columns; one block per row.
// ---------------------------------------------------------------------------
__global__ __launch_bounds__(256) void softmax_kernel(float* __restrict__ out) {
  __shared__ float red[256];
  const int t = threadIdx.x;
  float* p = out + (size_t)blockIdx.x * S_DIM;

  float v[8];
  float mx = -3.0e38f;
#pragma unroll
  for (int i = 0; i < 8; ++i) { v[i] = p[t + 256 * i]; mx = fmaxf(mx, v[i]); }
  red[t] = mx; __syncthreads();
  for (int s = 128; s > 0; s >>= 1) {
    if (t < s) red[t] = fmaxf(red[t], red[t + s]);
    __syncthreads();
  }
  mx = red[0]; __syncthreads();

  float sum = 0.0f;
#pragma unroll
  for (int i = 0; i < 8; ++i) { v[i] = __expf(v[i] - mx); sum += v[i]; }
  red[t] = sum; __syncthreads();
  for (int s = 128; s > 0; s >>= 1) {
    if (t < s) red[t] += red[t + s];
    __syncthreads();
  }
  float inv = 1.0f / red[0];
#pragma unroll
  for (int i = 0; i < 8; ++i) p[t + 256 * i] = v[i] * inv;
}

// ---------------------------------------------------------------------------
extern "C" void kernel_launch(void* const* d_in, const int* in_sizes, int n_in,
                              void* d_out, int out_size, void* d_ws, size_t ws_size,
                              hipStream_t stream) {
  (void)in_sizes; (void)n_in; (void)out_size; (void)ws_size;
  const float* x    = (const float*)d_in[0];   // [8, 2048, 2048]
  const float* w_ih = (const float*)d_in[1];   // [8192, 2048]
  // d_in[2] (w_hh) is mathematically dead: h0 = 0.
  float* out = (float*)d_out;                  // [8, 2048, 2048]

  unsigned short* xb = (unsigned short*)d_ws;             // 64 MiB bf16 x
  unsigned short* wb = xb + (size_t)M_TOTAL * S_DIM;      // 24 MiB bf16 w (i,g,o)

  cvt_x_kernel<<<M_TOTAL * S_DIM / (256 * 8), 256, 0, stream>>>(x, xb);
  cvt_w_kernel<<<3 * S_DIM * S_DIM / (256 * 8), 256, 0, stream>>>(w_ih, wb);
  dim3 grid(M_TOTAL / MT_BLK, S_DIM / NT_BLK);            // (256, 32)
  gemm_gates_kernel<<<grid, 256, 0, stream>>>(xb, wb, out);
  softmax_kernel<<<M_TOTAL, 256, 0, stream>>>(out);
}
